// Embed_74603581932034
// MI455X (gfx1250) — matrix operand order repaired and checked by reading, not compiled, not run on hardware
//
#include <hip/hip_runtime.h>
#include <hip/hip_bf16.h>

// ---------------------------------------------------------------------------
// LSTM LM on MI455X (gfx1250, wave32, WMMA).
//   B=32, T=64 (63 steps), E=512, H=1024 (4H=4096), V=32000
// Pipeline:
//   1) cvt weights f32->bf16 (W_ih, W_hh, fc_W)
//   2) embed gather -> X bf16 [2016,512]
//   3) xg = X @ W_ih^T + (b_ih+b_hh)   (WMMA bf16, f32 acc) [2016,4096]
//   4) 63x lstm_step: gates = xg[t] + h @ W_hh^T (WMMA), cell update,
//      write hs (f32 out + bf16 ws), ping-pong h
//   5) hT copy
//   6) logits = hs @ fc_W^T + fc_b (WMMA, 4x M-register-blocked),
//      stored transposed [B,V,63]
// ---------------------------------------------------------------------------

#define BATCH 32
#define TFULL 64
#define TT    63
#define EDIM  512
#define HDIM  1024
#define G4    4096
#define VDIM  32000

typedef __attribute__((ext_vector_type(16))) __bf16          v16bf;
typedef __attribute__((ext_vector_type(8)))  float           v8f;
typedef __attribute__((ext_vector_type(8)))  unsigned short  u16x8;
typedef __attribute__((ext_vector_type(16))) unsigned short  u16x16;

__device__ __forceinline__ unsigned short f2bf(float x) {
  unsigned int u = __builtin_bit_cast(unsigned int, x);
  unsigned int r = u + 0x7FFFu + ((u >> 16) & 1u);   // round-to-nearest-even
  return (unsigned short)(r >> 16);
}

__device__ __forceinline__ v16bf make_v16(u16x8 lo, u16x8 hi) {
  u16x16 t;
#pragma unroll
  for (int i = 0; i < 8; ++i) { t[i] = lo[i]; t[i + 8] = hi[i]; }
  return __builtin_bit_cast(v16bf, t);
}

// A-tile 16x32 bf16 from row-major [M,K]: lane = M%16,
// lanes 0-15 hold K = k0+{0..7,16..23}, lanes 16-31 hold K = k0+{8..15,24..31}
__device__ __forceinline__ v16bf load_a_tile(const unsigned short* __restrict__ A,
                                             int ldk, int m0, int k0, int lane) {
  int mr = m0 + (lane & 15);
  int kh = (lane & 16) ? 8 : 0;
  const unsigned short* p = A + (size_t)mr * ldk + k0 + kh;
  u16x8 lo = *(const u16x8*)(p);
  u16x8 hi = *(const u16x8*)(p + 16);
  return make_v16(lo, hi);
}

// B-tile 32x16 bf16 from row-major weight [N,K]: lane = N%16,
// lanes 0-15 hold K = k0+{0..7,16..23}, lanes 16-31 hold K = k0+{8..15,24..31}
__device__ __forceinline__ v16bf load_b_tile(const unsigned short* __restrict__ W,
                                             int ldk, int n0, int k0, int lane) {
  int nr = n0 + (lane & 15);
  int kb = k0 + ((lane & 16) ? 8 : 0);
  const unsigned short* p = W + (size_t)nr * ldk + kb;
  u16x8 lo = *(const u16x8*)(p);
  u16x8 hi = *(const u16x8*)(p + 16);
  return make_v16(lo, hi);
}

#define WMMA_BF16(acc, a, b)                                                   \
  (acc) = __builtin_amdgcn_wmma_f32_16x16x32_bf16(false, (a), false, (b),      \
                                                  (short)0, (acc), false, false)

// ------------------------------- kernels -----------------------------------

__global__ void cvt_bf16(const float* __restrict__ src,
                         unsigned short* __restrict__ dst, int n) {
  int i = blockIdx.x * blockDim.x + threadIdx.x;
  if (i < n) dst[i] = f2bf(src[i]);
}

__global__ void embed_gather(const int* __restrict__ sent,
                             const float* __restrict__ emb,
                             unsigned short* __restrict__ Xbf) {
  int r = blockIdx.x;                    // 0..2015 = b*63 + t
  int b = r / TT, t = r % TT;
  int tok = sent[b * TFULL + t];
  const float* src = emb + (size_t)tok * EDIM;
  unsigned short* dst = Xbf + (size_t)r * EDIM;
  for (int e = threadIdx.x; e < EDIM; e += blockDim.x) dst[e] = f2bf(src[e]);
}

// xg[m,n] = sum_k X[m,k]*W_ih[n,k] + b_ih[n] + b_hh[n]
__global__ void lstm_xgemm(const unsigned short* __restrict__ Xbf,
                           const unsigned short* __restrict__ Wih,
                           const float* __restrict__ bih,
                           const float* __restrict__ bhh,
                           float* __restrict__ xg) {
  int lane = threadIdx.x & 31, w = threadIdx.x >> 5;
  int n0 = blockIdx.x * 64 + (w & 3) * 16;     // 64 blocks * 64 = 4096
  int m0 = blockIdx.y * 32 + (w >> 2) * 16;    // 63 blocks * 32 = 2016
  v8f acc = {};
  for (int k0 = 0; k0 < EDIM; k0 += 32) {
    v16bf a = load_a_tile(Xbf, EDIM, m0, k0, lane);
    v16bf b = load_b_tile(Wih, EDIM, n0, k0, lane);
    WMMA_BF16(acc, a, b);
  }
  int nc = n0 + (lane & 15);
  float bias = bih[nc] + bhh[nc];
  int mbase = m0 + ((lane & 16) ? 8 : 0);
#pragma unroll
  for (int r = 0; r < 8; ++r)
    xg[(size_t)(mbase + r) * G4 + nc] = acc[r] + bias;
}

__global__ void init_state(unsigned short* __restrict__ h0,
                           unsigned short* __restrict__ h1,
                           float* __restrict__ c) {
  int i = blockIdx.x * blockDim.x + threadIdx.x;
  if (i < BATCH * HDIM) { h0[i] = 0; h1[i] = 0; c[i] = 0.0f; }
}

// One time step. Block = 8 waves: wave w -> gate (w>>1), m-tile (w&1).
// Each block owns hidden columns [blockIdx.x*16, +16) exclusively for c/h.
__global__ void lstm_step(const unsigned short* __restrict__ hin,   // [32,1024]
                          unsigned short* __restrict__ hout,        // [32,1024]
                          const unsigned short* __restrict__ Whh,   // [4096,1024]
                          const float* __restrict__ xg,             // [2016,4096]
                          float* __restrict__ cst,                  // [32,1024]
                          float* __restrict__ hs_f32,               // out: [32,63,1024]
                          unsigned short* __restrict__ hs_bf,       // [2048,1024]
                          int t) {
  __shared__ float lds_g[4 * 32 * 16];           // [gate][m][16 cols]
  int lane = threadIdx.x & 31, w = threadIdx.x >> 5;
  int gate = w >> 1, mt = w & 1;
  int m0 = mt * 16;
  int n0 = gate * HDIM + blockIdx.x * 16;
  v8f acc = {};
  for (int k0 = 0; k0 < HDIM; k0 += 32) {
    v16bf a = load_a_tile(hin, HDIM, m0, k0, lane);
    v16bf b = load_b_tile(Whh, HDIM, n0, k0, lane);
    WMMA_BF16(acc, a, b);
  }
  int c16 = lane & 15;
  int mbase = m0 + ((lane & 16) ? 8 : 0);
#pragma unroll
  for (int r = 0; r < 8; ++r) {
    int mm = mbase + r;
    float g = acc[r] + xg[(size_t)(mm * TT + t) * G4 + n0 + c16];
    lds_g[gate * 512 + mm * 16 + c16] = g;
  }
  __syncthreads();
  for (int e = threadIdx.x; e < 512; e += 256) {   // 32 rows x 16 cols
    int mm = e >> 4, cc = e & 15;
    int hcol = blockIdx.x * 16 + cc;
    float gi = lds_g[0 * 512 + e];
    float gf = lds_g[1 * 512 + e];
    float gg = lds_g[2 * 512 + e];
    float go = lds_g[3 * 512 + e];
    float si = 1.0f / (1.0f + __expf(-gi));
    float sf = 1.0f / (1.0f + __expf(-gf));
    float tg = tanhf(gg);
    float so = 1.0f / (1.0f + __expf(-go));
    size_t cidx = (size_t)mm * HDIM + hcol;
    float cn = sf * cst[cidx] + si * tg;
    float h  = so * tanhf(cn);
    cst[cidx] = cn;
    size_t hrow = (size_t)mm * TT + t;
    hs_f32[hrow * HDIM + hcol] = h;
    unsigned short hb = f2bf(h);
    hs_bf[hrow * HDIM + hcol] = hb;
    hout[cidx] = hb;
  }
}

__global__ void copy_hT(const float* __restrict__ hs_f32, float* __restrict__ hT) {
  int i = blockIdx.x * blockDim.x + threadIdx.x;
  if (i < BATCH * HDIM) {
    int b = i >> 10, j = i & (HDIM - 1);
    hT[i] = hs_f32[((size_t)b * TT + (TT - 1)) * HDIM + j];
  }
}

// logits[b,v,t] = sum_k hs[b,t,k]*fc_W[v,k] + fc_b[v], stored transposed.
// One wave computes ALL four t-tiles (63 timesteps) of batch b against one
// 16-wide v panel: B panel loaded once per wave, reused 4x (cuts L2 B-read
// traffic 4x and yields 4 WMMA per 10 b128 loads). Per lane: fixed v,
// 8 consecutive t values per acc register -> contiguous stores.
__global__ void fc_logits(const unsigned short* __restrict__ hsbf,  // [2048,1024]
                          const unsigned short* __restrict__ fcW,   // [32000,1024]
                          const float* __restrict__ fcb,
                          float* __restrict__ out) {               // [32,32000,63]
  int lane = threadIdx.x & 31, w = threadIdx.x >> 5;
  int b  = blockIdx.y;
  int v0 = blockIdx.x * 128 + w * 16;
  int r0 = b * TT;                    // partial tile reads padded rows, discarded
  __builtin_prefetch(fcW + (size_t)(v0 + (lane & 15)) * HDIM, 0, 1);
  v8f acc0 = {}, acc1 = {}, acc2 = {}, acc3 = {};
  for (int k0 = 0; k0 < HDIM; k0 += 32) {
    v16bf bm = load_b_tile(fcW, HDIM, v0, k0, lane);
    v16bf a0 = load_a_tile(hsbf, HDIM, r0 +  0, k0, lane);
    v16bf a1 = load_a_tile(hsbf, HDIM, r0 + 16, k0, lane);
    v16bf a2 = load_a_tile(hsbf, HDIM, r0 + 32, k0, lane);
    v16bf a3 = load_a_tile(hsbf, HDIM, r0 + 48, k0, lane);
    WMMA_BF16(acc0, a0, bm);
    WMMA_BF16(acc1, a1, bm);
    WMMA_BF16(acc2, a2, bm);
    WMMA_BF16(acc3, a3, bm);
  }
  int v = v0 + (lane & 15);
  float bias = fcb[v];
  int tb = (lane & 16) ? 8 : 0;
  float* op = out + ((size_t)b * VDIM + v) * TT;
#pragma unroll
  for (int r = 0; r < 8; ++r) op[ 0 + tb + r] = acc0[r] + bias;
#pragma unroll
  for (int r = 0; r < 8; ++r) op[16 + tb + r] = acc1[r] + bias;
#pragma unroll
  for (int r = 0; r < 8; ++r) op[32 + tb + r] = acc2[r] + bias;
#pragma unroll
  for (int r = 0; r < 8; ++r) {
    int t = 48 + tb + r;
    if (t < TT) op[t] = acc3[r] + bias;
  }
}

// ------------------------------- launcher ----------------------------------

extern "C" void kernel_launch(void* const* d_in, const int* in_sizes, int n_in,
                              void* d_out, int out_size, void* d_ws, size_t ws_size,
                              hipStream_t stream) {
  (void)in_sizes; (void)n_in; (void)out_size; (void)ws_size;
  const int*   sentence = (const int*)d_in[0];
  /* d_in[1] = lengths: unused by reference */
  const float* emb   = (const float*)d_in[2];
  const float* W_ih  = (const float*)d_in[3];
  const float* W_hh  = (const float*)d_in[4];
  const float* b_ih  = (const float*)d_in[5];
  const float* b_hh  = (const float*)d_in[6];
  const float* fc_W  = (const float*)d_in[7];
  const float* fc_b  = (const float*)d_in[8];

  float* out_logits = (float*)d_out;                                  // [32,32000,63]
  float* out_hs     = out_logits + (size_t)BATCH * VDIM * TT;         // [32,63,1024]
  float* out_hT     = out_hs + (size_t)BATCH * TT * HDIM;             // [32,1024]

  char* ws = (char*)d_ws;
  size_t off = 0;
  auto alloc = [&](size_t bytes) -> void* {
    void* p = ws + off;
    off = (off + bytes + 255) & ~(size_t)255;
    return p;
  };
  unsigned short* Xbf   = (unsigned short*)alloc((size_t)2016 * EDIM * 2);
  unsigned short* Wihb  = (unsigned short*)alloc((size_t)G4 * EDIM * 2);
  unsigned short* Whhb  = (unsigned short*)alloc((size_t)G4 * HDIM * 2);
  unsigned short* fcWb  = (unsigned short*)alloc((size_t)VDIM * HDIM * 2);
  float*          xg    = (float*)alloc((size_t)2016 * G4 * 4);
  unsigned short* hsbf  = (unsigned short*)alloc((size_t)2048 * HDIM * 2);  // padded
  unsigned short* hbuf0 = (unsigned short*)alloc((size_t)BATCH * HDIM * 2);
  unsigned short* hbuf1 = (unsigned short*)alloc((size_t)BATCH * HDIM * 2);
  float*          cst   = (float*)alloc((size_t)BATCH * HDIM * 4);

  cvt_bf16<<<(G4 * EDIM + 255) / 256, 256, 0, stream>>>(W_ih, Wihb, G4 * EDIM);
  cvt_bf16<<<(G4 * HDIM + 255) / 256, 256, 0, stream>>>(W_hh, Whhb, G4 * HDIM);
  cvt_bf16<<<(VDIM * HDIM + 255) / 256, 256, 0, stream>>>(fc_W, fcWb, VDIM * HDIM);
  embed_gather<<<2016, 128, 0, stream>>>(sentence, emb, Xbf);
  lstm_xgemm<<<dim3(64, 63), 256, 0, stream>>>(Xbf, Wihb, b_ih, b_hh, xg);
  init_state<<<(BATCH * HDIM + 255) / 256, 256, 0, stream>>>(hbuf0, hbuf1, cst);

  for (int t = 0; t < TT; ++t) {
    const unsigned short* hin = (t & 1) ? hbuf1 : hbuf0;
    unsigned short*      hout = (t & 1) ? hbuf0 : hbuf1;
    lstm_step<<<64, 256, 0, stream>>>(hin, hout, Whhb, xg, cst, out_hs, hsbf, t);
  }

  copy_hT<<<(BATCH * HDIM + 255) / 256, 256, 0, stream>>>(out_hs, out_hT);
  fc_logits<<<dim3(VDIM / 128, BATCH), 256, 0, stream>>>(hsbf, fcWb, fc_b, out_logits);
}
